// SequenceLearner_GLNN_14491219656705
// MI455X (gfx1250) — matrix operand order repaired and checked by reading, not compiled, not run on hardware
//
#include <hip/hip_runtime.h>
#include <hip/hip_bf16.h>

#define B_ 64
#define T_ 512
#define F_ 128
#define N_ 256
#define O_ 64
#define UNF 6
#define EPS_ 1e-8f

typedef __attribute__((ext_vector_type(16))) __bf16 v16bf;
typedef __attribute__((ext_vector_type(8)))  float  v8f;

__device__ __forceinline__ float fast_sigmoid(float z) {
    // v_exp_f32 + v_rcp_f32: transcendental path, co-executes with FMA stream
    return __builtin_amdgcn_rcpf(1.0f + __expf(-z));
}

// ---------------- param folding: {sigma, -sigma*mu, w*mask, w*mask*erev} ----------------
__global__ void ltc_prep(const float* __restrict__ ssig, const float* __restrict__ smu,
                         const float* __restrict__ sw,   const float* __restrict__ serev,
                         const float* __restrict__ smask,
                         const float* __restrict__ rsig, const float* __restrict__ rmu,
                         const float* __restrict__ rw,   const float* __restrict__ rerev,
                         const float* __restrict__ rmask,
                         float4* __restrict__ senpk, float4* __restrict__ recpk,
                         int SU, int UU) {
    int i = blockIdx.x * blockDim.x + threadIdx.x;
    if (i < SU) {
        float sg = ssig[i], mu = smu[i], w = sw[i] * smask[i];
        senpk[i] = make_float4(sg, -sg * mu, w, w * serev[i]);
    }
    if (i < UU) {
        float sg = rsig[i], mu = rmu[i], w = rw[i] * rmask[i];
        recpk[i] = make_float4(sg, -sg * mu, w, w * rerev[i]);
    }
}

// ---- fused 2-layer LTC scan: one 512-thread workgroup per batch element ----
// Source-dim reduction is split: 4 threads/unit for layer1 (32 srcs each),
// 2 threads/unit for layer2 (128 srcs each); partials combined via LDS.
__global__ __launch_bounds__(512, 1) void ltc_scan(
    const float* __restrict__ x,
    const float* __restrict__ iw1p, const float* __restrict__ ib1p,
    const float* __restrict__ gl1p, const float* __restrict__ vl1p, const float* __restrict__ cm1p,
    const float* __restrict__ ow1p, const float* __restrict__ ob1p,
    const float* __restrict__ iw2p, const float* __restrict__ ib2p,
    const float* __restrict__ gl2p, const float* __restrict__ vl2p, const float* __restrict__ cm2p,
    const float* __restrict__ ow2p, const float* __restrict__ ob2p,
    const float4* __restrict__ sen1, const float4* __restrict__ rec1,
    const float4* __restrict__ sen2, const float4* __restrict__ rec2,
    float* __restrict__ h2buf)
{
    const int b   = blockIdx.x;
    const int tid = threadIdx.x;          // 0..511
    const int u1  = tid & (F_ - 1);       // layer-1 unit, 4 threads each
    const int q1  = tid >> 7;             // layer-1 source-quarter 0..3
    const int u2  = tid & (N_ - 1);       // layer-2 unit, 2 threads each
    const int h2i = tid >> 8;             // layer-2 source-half 0..1

    __shared__ __attribute__((aligned(16))) float v1s[F_];
    __shared__ __attribute__((aligned(16))) float i1s[F_];
    __shared__ __attribute__((aligned(16))) float v2s[N_];
    __shared__ __attribute__((aligned(16))) float i2s[F_];
    __shared__ __attribute__((aligned(16))) float pn[512];
    __shared__ __attribute__((aligned(16))) float pd[512];

    // per-thread constants (owner threads only)
    float iw1 = 0.f, ib1 = 0.f, ow1 = 0.f, ob1 = 0.f, iw2 = 0.f, ib2 = 0.f;
    float cmt1 = 0.f, gl1 = 0.f, glvl1 = 0.f;
    if (tid < F_) {                       // layer-1 owners (q1 == 0)
        iw1 = iw1p[tid]; ib1 = ib1p[tid];
        ow1 = ow1p[tid]; ob1 = ob1p[tid];
        iw2 = iw2p[tid]; ib2 = ib2p[tid];
        cmt1 = cm1p[tid] * (float)UNF;
        gl1  = gl1p[tid]; glvl1 = gl1 * vl1p[tid];
    }
    float cmt2 = 0.f, gl2 = 0.f, glvl2 = 0.f;
    if (tid < N_) {                       // layer-2 owners (h2i == 0)
        cmt2 = cm2p[tid] * (float)UNF;
        gl2  = gl2p[tid]; glvl2 = gl2 * vl2p[tid];
    }
    float ow2 = 0.f, ob2 = 0.f;
    if (tid < O_) { ow2 = ow2p[tid]; ob2 = ob2p[tid]; }

    if (tid < F_) v1s[tid] = 0.f;
    if (tid < N_) v2s[tid] = 0.f;
    float vc1 = 0.f, vc2 = 0.f;
    __syncthreads();

    const int s1_lo = q1 * (F_ / 4);                // layer-1 recurrent range (32)
    const int s1s_lo = q1 * (F_ / 4);               // layer-1 sensory range  (32)
    const int s2_lo = h2i * (N_ / 2);               // layer-2 recurrent range (128)
    const int s2s_lo = h2i * (F_ / 2);              // layer-2 sensory range   (64)

    for (int t = 0; t < T_; ++t) {
        if (tid < F_) i1s[tid] = x[((size_t)b * T_ + t) * F_ + tid] * iw1 + ib1;
        __syncthreads();

        // ---- layer-1 sensory partial (this quarter's share) ----
        float ns1 = 0.f, ds1 = 0.f;
        for (int s = s1s_lo; s < s1s_lo + F_ / 4; s += 4) {
            const float4 iv = *reinterpret_cast<const float4*>(&i1s[s]);
            const float* ivp = &iv.x;
            #pragma unroll
            for (int j = 0; j < 4; ++j) {
                const float4 p = sen1[(s + j) * F_ + u1];
                const float sg = fast_sigmoid(fmaf(p.x, ivp[j], p.y));
                ns1 = fmaf(p.w, sg, ns1);
                ds1 = fmaf(p.z, sg, ds1);
            }
        }
        // ---- layer-1 ODE unfolds ----
        #pragma unroll 1
        for (int k = 0; k < UNF; ++k) {
            float wn = ns1, wd = ds1;
            for (int s = s1_lo; s < s1_lo + F_ / 4; s += 4) {
                const float4 vv = *reinterpret_cast<const float4*>(&v1s[s]);
                const float* vp = &vv.x;
                #pragma unroll
                for (int j = 0; j < 4; ++j) {
                    const float4 p = rec1[(s + j) * F_ + u1];
                    const float sg = fast_sigmoid(fmaf(p.x, vp[j], p.y));
                    wn = fmaf(p.w, sg, wn);
                    wd = fmaf(p.z, sg, wd);
                }
            }
            pn[tid] = wn; pd[tid] = wd;
            __syncthreads();
            if (tid < F_) {
                const float wnT = (pn[tid] + pn[tid + F_]) + (pn[tid + 2 * F_] + pn[tid + 3 * F_]);
                const float wdT = (pd[tid] + pd[tid + F_]) + (pd[tid + 2 * F_] + pd[tid + 3 * F_]);
                vc1 = (cmt1 * vc1 + glvl1 + wnT) / (cmt1 + gl1 + wdT + EPS_);
                v1s[tid] = vc1;
            }
            __syncthreads();
        }
        if (tid < F_) i2s[tid] = (vc1 * ow1 + ob1) * iw2 + ib2;
        __syncthreads();

        // ---- layer-2 sensory partial (this half's share) ----
        float ns2 = 0.f, ds2 = 0.f;
        for (int s = s2s_lo; s < s2s_lo + F_ / 2; s += 4) {
            const float4 iv = *reinterpret_cast<const float4*>(&i2s[s]);
            const float* ivp = &iv.x;
            #pragma unroll
            for (int j = 0; j < 4; ++j) {
                const float4 p = sen2[(s + j) * N_ + u2];
                const float sg = fast_sigmoid(fmaf(p.x, ivp[j], p.y));
                ns2 = fmaf(p.w, sg, ns2);
                ds2 = fmaf(p.z, sg, ds2);
            }
        }
        // ---- layer-2 ODE unfolds ----
        #pragma unroll 1
        for (int k = 0; k < UNF; ++k) {
            float wn = ns2, wd = ds2;
            for (int s = s2_lo; s < s2_lo + N_ / 2; s += 4) {
                const float4 vv = *reinterpret_cast<const float4*>(&v2s[s]);
                const float* vp = &vv.x;
                #pragma unroll
                for (int j = 0; j < 4; ++j) {
                    const float4 p = rec2[(s + j) * N_ + u2];
                    const float sg = fast_sigmoid(fmaf(p.x, vp[j], p.y));
                    wn = fmaf(p.w, sg, wn);
                    wd = fmaf(p.z, sg, wd);
                }
            }
            pn[tid] = wn; pd[tid] = wd;
            __syncthreads();
            if (tid < N_) {
                const float wnT = pn[tid] + pn[tid + N_];
                const float wdT = pd[tid] + pd[tid + N_];
                vc2 = (cmt2 * vc2 + glvl2 + wnT) / (cmt2 + gl2 + wdT + EPS_);
                v2s[tid] = vc2;
            }
            __syncthreads();
        }
        if (tid < O_) h2buf[((size_t)b * T_ + t) * O_ + tid] = vc2 * ow2 + ob2;
    }
}

// ---------------- final FC via bf16 WMMA: [32768,64] x [64,64]^T + bias ----------------
__global__ __launch_bounds__(256) void fc_wmma_kernel(
    const float* __restrict__ h2, const float* __restrict__ fcw,
    const float* __restrict__ fcb, float* __restrict__ out)
{
    const int lane = threadIdx.x & 31;
    const int wave = (blockIdx.x * blockDim.x + threadIdx.x) >> 5;
    const int tile_n = wave & 3;     // 64 / 16 = 4 N tiles
    const int tile_m = wave >> 2;    // (B*T)/16 = 2048 M tiles
    if (tile_m >= (B_ * T_) / 16) return;   // wave-uniform

    const int halfsel = (lane >= 16) ? 1 : 0;
    const int mr = tile_m * 16 + (lane & 15);   // A row
    const int nc = tile_n * 16 + (lane & 15);   // B column / D column

    v8f c = {};
    #pragma unroll
    for (int kh = 0; kh < 2; ++kh) {   // K = 64 -> two 16x16x32 WMMAs
        v16bf a, bm;
        #pragma unroll
        for (int e = 0; e < 16; ++e) {
            const int j = e >> 1;
            const int i = e & 1;
            // A (16-bit, 16x32) lane layout per ISA: K = [0..7|16..23] + 8*halfsel
            const int ka = ((j >= 4) ? 16 : 0) + 8 * halfsel + (j & 3) * 2 + i + kh * 32;
            // B (16-bit, 32x16) lane layout per ISA: K = [0..7|16..23] + 8*halfsel
            const int kb = ((j >= 4) ? 16 : 0) + 8 * halfsel + (j & 3) * 2 + i + kh * 32;
            a[e]  = (__bf16)h2[(size_t)mr * O_ + ka];
            bm[e] = (__bf16)fcw[(size_t)nc * O_ + kb];
        }
        c = __builtin_amdgcn_wmma_f32_16x16x32_bf16(false, a, false, bm,
                                                    (short)0, c, false, false);
    }
    const float bias = fcb[nc];
    const int mbase = tile_m * 16 + halfsel * 8;
    #pragma unroll
    for (int r = 0; r < 8; ++r)
        out[(size_t)(mbase + r) * O_ + nc] = c[r] + bias;
}

extern "C" void kernel_launch(void* const* d_in, const int* in_sizes, int n_in,
                              void* d_out, int out_size, void* d_ws, size_t ws_size,
                              hipStream_t stream) {
    (void)in_sizes; (void)n_in; (void)out_size; (void)ws_size;
    const float* x = (const float*)d_in[0];
    // l1 params (indices 1..17)
    const float* l1_iw  = (const float*)d_in[1];
    const float* l1_ib  = (const float*)d_in[2];
    const float* l1_sw  = (const float*)d_in[3];
    const float* l1_ssg = (const float*)d_in[4];
    const float* l1_smu = (const float*)d_in[5];
    const float* l1_ser = (const float*)d_in[6];
    const float* l1_smk = (const float*)d_in[7];
    const float* l1_w   = (const float*)d_in[8];
    const float* l1_sg  = (const float*)d_in[9];
    const float* l1_mu  = (const float*)d_in[10];
    const float* l1_er  = (const float*)d_in[11];
    const float* l1_mk  = (const float*)d_in[12];
    const float* l1_gl  = (const float*)d_in[13];
    const float* l1_vl  = (const float*)d_in[14];
    const float* l1_cm  = (const float*)d_in[15];
    const float* l1_ow  = (const float*)d_in[16];
    const float* l1_ob  = (const float*)d_in[17];
    // l2 params (indices 18..34)
    const float* l2_iw  = (const float*)d_in[18];
    const float* l2_ib  = (const float*)d_in[19];
    const float* l2_sw  = (const float*)d_in[20];
    const float* l2_ssg = (const float*)d_in[21];
    const float* l2_smu = (const float*)d_in[22];
    const float* l2_ser = (const float*)d_in[23];
    const float* l2_smk = (const float*)d_in[24];
    const float* l2_w   = (const float*)d_in[25];
    const float* l2_sg  = (const float*)d_in[26];
    const float* l2_mu  = (const float*)d_in[27];
    const float* l2_er  = (const float*)d_in[28];
    const float* l2_mk  = (const float*)d_in[29];
    const float* l2_gl  = (const float*)d_in[30];
    const float* l2_vl  = (const float*)d_in[31];
    const float* l2_cm  = (const float*)d_in[32];
    const float* l2_ow  = (const float*)d_in[33];
    const float* l2_ob  = (const float*)d_in[34];
    const float* fcw    = (const float*)d_in[35];
    const float* fcb    = (const float*)d_in[36];

    // workspace layout (floats): h2 | l1_sen | l1_rec | l2_sen | l2_rec
    float*  ws     = (float*)d_ws;
    float*  h2buf  = ws;                                   // 64*512*64   = 2097152
    float4* l1_sen = (float4*)(ws + 2097152);              // 128*128 f4  =   65536 f
    float4* l1_rec = (float4*)(ws + 2097152 + 65536);      // 128*128 f4  =   65536 f
    float4* l2_sen = (float4*)(ws + 2097152 + 131072);     // 128*256 f4  =  131072 f
    float4* l2_rec = (float4*)(ws + 2097152 + 262144);     // 256*256 f4  =  262144 f

    // 1) fold params
    ltc_prep<<<(F_ * F_ + 255) / 256, 256, 0, stream>>>(
        l1_ssg, l1_smu, l1_sw, l1_ser, l1_smk,
        l1_sg,  l1_mu,  l1_w,  l1_er,  l1_mk,
        l1_sen, l1_rec, F_ * F_, F_ * F_);
    ltc_prep<<<(N_ * N_ + 255) / 256, 256, 0, stream>>>(
        l2_ssg, l2_smu, l2_sw, l2_ser, l2_smk,
        l2_sg,  l2_mu,  l2_w,  l2_er,  l2_mk,
        l2_sen, l2_rec, F_ * N_, N_ * N_);

    // 2) fused two-layer LTC scan (one 512-thread workgroup per batch element)
    ltc_scan<<<B_, 512, 0, stream>>>(
        x,
        l1_iw, l1_ib, l1_gl, l1_vl, l1_cm, l1_ow, l1_ob,
        l2_iw, l2_ib, l2_gl, l2_vl, l2_cm, l2_ow, l2_ob,
        l1_sen, l1_rec, l2_sen, l2_rec,
        h2buf);

    // 3) final FC via WMMA: 2048 m-tiles * 4 n-tiles = 8192 waves = 1024 blocks
    fc_wmma_kernel<<<1024, 256, 0, stream>>>(h2buf, fcw, fcb, (float*)d_out);
}
